// QuantLinearMatVec_88854283420079
// MI455X (gfx1250) — matrix operand order repaired and checked by reading, not compiled, not run on hardware
//
#include <hip/hip_runtime.h>
#include <hip/hip_bf16.h>
#include <hip/hip_fp16.h>

// ---------------------------------------------------------------------------
// Quantized linear:  Y = bf16( fp16( bf16(X) @ bf16(W)^T ) + fp16(bias) )
//   X: [B, K] fp32, W: [N, K] fp32, bias: [N] fp32, Y: [B, N] bf16
//   B=2048, K=4096, N=4096 (derived from in_sizes; assumed tile-divisible)
//
// Tiling: block tile 128x128, BK=32, 256 threads = 8 wave32.
// Wave grid 4(M) x 2(N): each wave computes 32x64 = 2x4 WMMA 16x16 tiles.
// fp32 -> bf16 conversion (the reference's quant()) happens during the
// global->LDS staging, so the WMMA inner loop runs pure bf16.
//
// Double-buffered LDS: one workgroup barrier per k-step; the cvt+ds_store of
// tile k+1 overlaps the 8 WMMAs of tile k (16-bit WMMA co-executes with
// VALU/LDS per CDNA5 scheduling rules).
// ---------------------------------------------------------------------------

typedef __attribute__((ext_vector_type(16))) __bf16 v16bf;
typedef __attribute__((ext_vector_type(8)))  __bf16 v8bf;
typedef __attribute__((ext_vector_type(4)))  __bf16 v4bf;
typedef __attribute__((ext_vector_type(8)))  float  v8f;

#define BM 128
#define BN 128
#define BK 32
#define LDSS 40   // padded row stride in bf16 elems (80B = 20 banks -> conflict-free b128)

union FragU { v16bf v; v8bf h[2]; };

__global__ __launch_bounds__(256) void qlinear_bf16_wmma(
    const float* __restrict__ X,
    const float* __restrict__ W,
    const float* __restrict__ bias,
    __hip_bfloat16* __restrict__ Y,
    int K, int N)
{
  // double-buffered staging tiles (2 * (10240 + 10240) B = 40 KB of 320 KB/WGP)
  __shared__ alignas(16) __bf16 As[2][BM * LDSS];
  __shared__ alignas(16) __bf16 Bs[2][BN * LDSS];

  const int tid  = threadIdx.x;
  const int lane = tid & 31;
  const int wave = tid >> 5;
  const int wm   = wave & 3;    // wave M offset = wm*32
  const int wn   = wave >> 2;   // wave N offset = wn*64

  const long blockM = (long)blockIdx.y * BM;
  const long blockN = (long)blockIdx.x * BN;

  // cooperative loader: tid -> (row 0..31 (+it*32), col*4), one float4 each
  const int lr = tid >> 3;
  const int lc = (tid & 7) * 4;

  // WMMA fragment addressing (ISA 7.12.2 layouts, wave32)
  const int frow = lane & 15;          // matrix row (A) / B column index
  const int ka   = (lane >> 4) * 8;    // A: 8-elem chunks at ka and ka+16
  const int kb   = (lane >> 4) * 16;   // unused: B also reads 8-elem chunks at ka and ka+16

  v8f acc[2][4];
#pragma unroll
  for (int i = 0; i < 2; ++i)
#pragma unroll
    for (int j = 0; j < 4; ++j)
      acc[i][j] = (v8f)0.0f;

  // ---- load k-tile 0 and stage into LDS buffer 0 ----------------------
  float4 pa[4], pb[4];
#pragma unroll
  for (int it = 0; it < 4; ++it) {
    const long r = lr + it * 32;
    pa[it] = *(const float4*)&X[(blockM + r) * (long)K + lc];
    pb[it] = *(const float4*)&W[(blockN + r) * (long)K + lc];
  }
#pragma unroll
  for (int it = 0; it < 4; ++it) {
    const int r = lr + it * 32;
    v4bf a4, b4;
    a4.x = (__bf16)pa[it].x; a4.y = (__bf16)pa[it].y;
    a4.z = (__bf16)pa[it].z; a4.w = (__bf16)pa[it].w;
    b4.x = (__bf16)pb[it].x; b4.y = (__bf16)pb[it].y;
    b4.z = (__bf16)pb[it].z; b4.w = (__bf16)pb[it].w;
    *(v4bf*)&As[0][r * LDSS + lc] = a4;
    *(v4bf*)&Bs[0][r * LDSS + lc] = b4;
  }
  __syncthreads();

  const int ksteps = K / BK;
  for (int ks = 0; ks < ksteps; ++ks) {
    const int buf  = ks & 1;
    const int nbuf = buf ^ 1;
    const bool have_next = (ks + 1) < ksteps;

    // ---- issue global loads for k-tile ks+1 first (latency hiding) -----
    if (have_next) {
      const int k0 = (ks + 1) * BK;
#pragma unroll
      for (int it = 0; it < 4; ++it) {
        const long r = lr + it * 32;
        pa[it] = *(const float4*)&X[(blockM + r) * (long)K + k0 + lc];
        pb[it] = *(const float4*)&W[(blockN + r) * (long)K + k0 + lc];
      }
    }

    // ---- fragments from LDS buffer `buf` and 8 WMMAs -------------------
    FragU af[2], bfr[4];
#pragma unroll
    for (int i = 0; i < 2; ++i) {
      const int row = wm * 32 + i * 16 + frow;
      af[i].h[0] = *(const v8bf*)&As[buf][row * LDSS + ka];
      af[i].h[1] = *(const v8bf*)&As[buf][row * LDSS + 16 + ka];
    }
#pragma unroll
    for (int j = 0; j < 4; ++j) {
      const int row = wn * 64 + j * 16 + frow;
      bfr[j].h[0] = *(const v8bf*)&Bs[buf][row * LDSS + ka];
      bfr[j].h[1] = *(const v8bf*)&Bs[buf][row * LDSS + 16 + ka];
    }
#pragma unroll
    for (int i = 0; i < 2; ++i)
#pragma unroll
      for (int j = 0; j < 4; ++j)
        acc[i][j] = __builtin_amdgcn_wmma_f32_16x16x32_bf16(
            /*neg_a=*/false, af[i].v,
            /*neg_b=*/false, bfr[j].v,
            /*c_mod=*/(short)0, acc[i][j],
            /*reuse_a=*/false, /*reuse_b=*/false);

    // ---- convert + store k-tile ks+1 into the other buffer -------------
    // Safe without a pre-barrier: nbuf was last READ in iteration ks-1,
    // and the barrier at the end of that iteration ordered those reads.
    if (have_next) {
#pragma unroll
      for (int it = 0; it < 4; ++it) {
        const int r = lr + it * 32;
        v4bf a4, b4;
        a4.x = (__bf16)pa[it].x; a4.y = (__bf16)pa[it].y;
        a4.z = (__bf16)pa[it].z; a4.w = (__bf16)pa[it].w;
        b4.x = (__bf16)pb[it].x; b4.y = (__bf16)pb[it].y;
        b4.z = (__bf16)pb[it].z; b4.w = (__bf16)pb[it].w;
        *(v4bf*)&As[nbuf][r * LDSS + lc] = a4;
        *(v4bf*)&Bs[nbuf][r * LDSS + lc] = b4;
      }
    }
    __syncthreads();   // single barrier per k-step
  }

  // ---- epilogue: fp16 quant of acc and bias, add, bf16 quant, store ----
  // C/D layout: VGPR r, lanes 0-15 -> (M=r, N=lane); lanes 16-31 -> (M=r+8, N=lane-16)
  const long col0 = blockN + wn * 64;
#pragma unroll
  for (int j = 0; j < 4; ++j) {
    const long col = col0 + j * 16 + (lane & 15);
    const float bq = __half2float(__float2half(bias[col]));   // quant(bias, fp16)
#pragma unroll
    for (int i = 0; i < 2; ++i) {
      const long row0 = blockM + wm * 32 + i * 16 + (lane >> 4) * 8;
#pragma unroll
      for (int r = 0; r < 8; ++r) {
        const float vq = __half2float(__float2half(acc[i][j][r]));  // quant(mm, fp16)
        Y[(row0 + r) * (long)N + col] = __float2bfloat16(vq + bq);  // quant(+, bf16)
      }
    }
  }
}

extern "C" void kernel_launch(void* const* d_in, const int* in_sizes, int n_in,
                              void* d_out, int out_size, void* d_ws, size_t ws_size,
                              hipStream_t stream) {
  const float* X    = (const float*)d_in[0];
  const float* W    = (const float*)d_in[1];
  const float* bias = (const float*)d_in[2];
  __hip_bfloat16* Y = (__hip_bfloat16*)d_out;

  const int N = in_sizes[2];            // DOUT (bias length)
  const int K = in_sizes[1] / N;        // DIN
  const int B = in_sizes[0] / K;        // batch

  dim3 grid(N / BN, B / BM);
  qlinear_bf16_wmma<<<grid, dim3(256), 0, stream>>>(X, W, bias, Y, K, N);
}
